// Reins_Attention4_55216099557939
// MI455X (gfx1250) — matrix operand order repaired and checked by reading, not compiled, not run on hardware
//
#include <hip/hip_runtime.h>
#include <hip/hip_bf16.h>

// ---------------------------------------------------------------------------
// Problem constants
// ---------------------------------------------------------------------------
#define EMBED   1024
#define TOKLEN  100
#define TOKP    128     // tokens padded to 128 (multiple of 32 for WMMA K)
#define TOKDIM  256
#define MLPDIM  128
#define SEQ     4097
#define BATCH   8
#define MROWS   (4096 * 8)          // body rows (flattened (seq-1, batch))
#define CLS_ELEMS (BATCH * EMBED)   // 8192 floats of the cls token row

#define LDSPAD  40                  // padded row stride (halfs): 80B = 20 banks
                                    // -> (col*20) mod 64 is a perfect bank
                                    //    permutation: conflict-free b128 loads

typedef __attribute__((ext_vector_type(16))) _Float16 v16h;
typedef __attribute__((ext_vector_type(8)))  _Float16 v8h;
typedef __attribute__((ext_vector_type(8)))  float    v8f;
typedef __attribute__((ext_vector_type(4)))  int      v4i;

// ---------------------------------------------------------------------------
// CDNA5 async global->LDS copy (ASYNCcnt) with portable fallback
// ---------------------------------------------------------------------------
#if defined(__has_builtin)
#if __has_builtin(__builtin_amdgcn_global_load_async_to_lds_b128)
#define HAVE_ASYNC_LDS 1
#endif
#if __has_builtin(__builtin_amdgcn_s_wait_asynccnt)
#define HAVE_WAIT_ASYNC 1
#endif
#endif

typedef __attribute__((address_space(1))) v4i* gv4i_p;   // global int4*
typedef __attribute__((address_space(3))) v4i* lv4i_p;   // LDS int4*

__device__ inline void async_cp16(const _Float16* g, _Float16* l) {
#ifdef HAVE_ASYNC_LDS
    __builtin_amdgcn_global_load_async_to_lds_b128(
        (gv4i_p)g, (lv4i_p)l, 0, 0);
#else
    *(v8h*)l = *(const v8h*)g;   // synchronous fallback
#endif
}

#ifdef HAVE_WAIT_ASYNC
#define WAIT_ASYNC(n) __builtin_amdgcn_s_wait_asynccnt(n)
#else
#define WAIT_ASYNC(n) asm volatile("s_wait_asynccnt " #n)
#endif

// ---------------------------------------------------------------------------
// WMMA helpers (wave32, v_wmma_f32_16x16x32_f16)
//
// A fragment (16x32, MxK): lane 0-15 -> row M=lane, K = 0..7 (v0..v3) and
//   K = 16..23 (v4..v7); lane 16-31 -> row M=lane-16, K = 8..15 / 24..31.
// B fragment (32x16, KxN): lane -> column N=lane&15, K = 0..7 / 16..23 (lanes 0-15)
//   or K = 8..15 / 24..31 (lanes 16-31): two 8-half K runs per lane when B is
//   stored transposed (N x K row-major).
// C/D: VGPR r, lane l -> (M = r + 8*(l>>4), N = l&15).
// ---------------------------------------------------------------------------
__device__ inline v8f wmma_f16(v16h a, v16h b, v8f c) {
    return __builtin_amdgcn_wmma_f32_16x16x32_f16(
        /*neg_a=*/false, a, /*neg_b=*/false, b,
        /*c_mod=*/(short)0, c, /*reuse_a=*/false, /*reuse_b=*/false);
}

// A fragment from a row-major fp32 matrix (converted to f16 on the fly).
__device__ inline v16h load_a_f32(const float* base, int stride, int lane) {
    int row = lane & 15;
    int kb  = (lane & 16) ? 8 : 0;
    const float* p = base + (long)row * stride + kb;
    v16h a;
#pragma unroll
    for (int i = 0; i < 8; ++i) a[i] = (_Float16)p[i];
#pragma unroll
    for (int i = 0; i < 8; ++i) a[8 + i] = (_Float16)p[16 + i];
    return a;
}

// A fragment from a row-major f16 matrix (two 16B loads; works for global
// stride 1024 and padded-LDS stride LDSPAD).
__device__ inline v16h load_a_h(const _Float16* base, int stride, int lane) {
    int row = lane & 15;
    int kb  = (lane & 16) ? 8 : 0;
    const _Float16* p = base + (long)row * stride + kb;
    v8h lo = *(const v8h*)(p);
    v8h hi = *(const v8h*)(p + 16);
    v16h a;
#pragma unroll
    for (int i = 0; i < 8; ++i) { a[i] = lo[i]; a[8 + i] = hi[i]; }
    return a;
}

// B fragment from an N x K row-major ("B transposed") f16 matrix.
// Two 16B loads so any 16B-aligned stride works (global or padded LDS).
__device__ inline v16h load_b_h(const _Float16* baseT, int stride, int lane) {
    int col = lane & 15;
    int kb  = (lane & 16) ? 8 : 0;
    const _Float16* p = baseT + (long)col * stride + kb;
    v8h lo = *(const v8h*)(p);
    v8h hi = *(const v8h*)(p + 16);
    v16h b;
#pragma unroll
    for (int i = 0; i < 8; ++i) { b[i] = lo[i]; b[8 + i] = hi[i]; }
    return b;
}

// ---------------------------------------------------------------------------
// Small prep kernels (scalar fp32; < 1% of total FLOPs)
// ---------------------------------------------------------------------------
__global__ __launch_bounds__(256) void k_copy_cls(const float* x, float* out) {
    int i = blockIdx.x * 256 + threadIdx.x;
    if (i < CLS_ELEMS) out[i] = x[i];
}

// H = gelu(B[layer] @ Wd^T + bd)   -> (1024, 128)
__global__ __launch_bounds__(256) void k_mlp1(const float* B, const int* layerp,
                                              const float* Wd, const float* bd,
                                              float* H) {
    int idx = blockIdx.x * 256 + threadIdx.x;           // 1024*128
    int i = idx >> 7, j = idx & 127;
    const float* Bl = B + (long)(*layerp) * EMBED * TOKDIM;
    float s = bd[j];
    const float* br = Bl + (long)i * TOKDIM;
    const float* wr = Wd + (long)j * TOKDIM;
#pragma unroll 4
    for (int k = 0; k < TOKDIM; ++k) s += br[k] * wr[k];
    H[idx] = 0.5f * s * (1.0f + erff(s * 0.70710678118654752f));  // exact gelu
}

// fB = H @ Wu^T + bu   -> (1024, 1024)
__global__ __launch_bounds__(256) void k_mlp2(const float* H, const float* Wu,
                                              const float* bu, float* fB) {
    int idx = blockIdx.x * 256 + threadIdx.x;           // 1024*1024
    int i = idx >> 10, j = idx & 1023;
    float s = bu[j];
    const float* hr = H + (long)i * MLPDIM;
    const float* wr = Wu + (long)j * MLPDIM;
#pragma unroll 4
    for (int k = 0; k < MLPDIM; ++k) s += hr[k] * wr[k];
    fB[idx] = s;
}

// tokens = A[0] @ fB  -> (100,1024); also write f16 copy padded to 128 rows
__global__ __launch_bounds__(256) void k_tokens(const float* A, const float* fB,
                                                float* tokF, _Float16* tokH) {
    int idx = blockIdx.x * 256 + threadIdx.x;           // 128*1024
    int m = idx >> 10, c = idx & 1023;
    if (m < TOKLEN) {
        float s = 0.f;
        const float* ar = A + (long)m * EMBED;
#pragma unroll 4
        for (int k = 0; k < EMBED; ++k) s += ar[k] * fB[(long)k * EMBED + c];
        tokF[idx] = s;
        tokH[idx] = (_Float16)s;
    } else {
        tokF[idx] = 0.f;
        tokH[idx] = (_Float16)0.f;
    }
}

// tfeat = tokens @ Wt^T + bt, stored transposed+padded as f16: (1024 x 128)
__global__ __launch_bounds__(256) void k_tfeat(const float* tokF, const float* Wt,
                                               const float* bt, _Float16* tfT) {
    int idx = blockIdx.x * 256 + threadIdx.x;           // 1024*128
    int m = idx & 127, c = idx >> 7;
    _Float16 v = (_Float16)0.f;
    if (m < TOKLEN) {
        float s = bt[c];
        const float* tr = tokF + (long)m * EMBED;
        const float* wr = Wt + (long)c * EMBED;
#pragma unroll 4
        for (int k = 0; k < EMBED; ++k) s += tr[k] * wr[k];
        v = (_Float16)s;
    }
    tfT[(long)c * TOKP + m] = v;
}

// f16 copy of Wf (Wf is (Nout x Kin) row-major == exactly the B^T layout)
__global__ __launch_bounds__(256) void k_cast(const float* Wf, _Float16* WfH) {
    int idx = blockIdx.x * 256 + threadIdx.x;           // 1024*1024
    WfH[idx] = (_Float16)Wf[idx];
}

// ---------------------------------------------------------------------------
// Fused cross-attention: 16 query rows per block (256 thr = 8 waves).
//   scores = body . tokens^T * EMBED^-0.5 ; softmax over 100 (padded 128)
//   delta  = attn . tfeat ;  U = (delta + body) as f16
// ---------------------------------------------------------------------------
__global__ __launch_bounds__(256) void k_attn(const float* x,
                                              const _Float16* tokH,
                                              const _Float16* tfT,
                                              _Float16* U) {
    __shared__ float     s_sc[16 * TOKP];   // fp32 scores / exp scratch
    __shared__ _Float16  s_at[16 * TOKP];   // f16 attention weights

    const float* body = x + CLS_ELEMS;      // skip cls row
    const int tid = threadIdx.x;
    const int w = tid >> 5, lane = tid & 31;
    const long qbase = (long)blockIdx.x * 16;

    // ---- phase 1: scores tile 16x128; wave w owns tokens [16w, 16w+16) ----
    v8f s = {};
    const float*    abase = body + qbase * EMBED;
    const _Float16* bbase = tokH + (long)(w * 16) * EMBED;
    for (int ks = 0; ks < EMBED / 32; ++ks) {
        v16h a = load_a_f32(abase + ks * 32, EMBED, lane);
        v16h b = load_b_h(bbase + ks * 32, EMBED, lane);
        s = wmma_f16(a, b, s);
    }
    {
        int rowhi = (lane >> 4) * 8;
        int col = w * 16 + (lane & 15);
#pragma unroll
        for (int r = 0; r < 8; ++r)
            s_sc[(rowhi + r) * TOKP + col] = s[r] * 0.03125f;  // 1/sqrt(1024)
    }
    __syncthreads();

    // ---- phase 2: softmax; 16 threads per row, 8 cols each ----
    {
        int row = tid >> 4, sub = tid & 15;
        float mx = -1e30f;
#pragma unroll
        for (int j = 0; j < 8; ++j) {
            int c = sub + 16 * j;
            if (c < TOKLEN) mx = fmaxf(mx, s_sc[row * TOKP + c]);
        }
        for (int d = 8; d >= 1; d >>= 1) mx = fmaxf(mx, __shfl_xor(mx, d, 16));
        float sum = 0.f;
#pragma unroll
        for (int j = 0; j < 8; ++j) {
            int c = sub + 16 * j;
            if (c < TOKLEN) {
                float e = __expf(s_sc[row * TOKP + c] - mx);
                s_sc[row * TOKP + c] = e;
                sum += e;
            }
        }
        for (int d = 8; d >= 1; d >>= 1) sum += __shfl_xor(sum, d, 16);
        float rs = 1.f / sum;
#pragma unroll
        for (int j = 0; j < 8; ++j) {
            int c = sub + 16 * j;
            s_at[row * TOKP + c] =
                (c < TOKLEN) ? (_Float16)(s_sc[row * TOKP + c] * rs)
                             : (_Float16)0.f;
        }
    }
    __syncthreads();

    // ---- phase 3: delta = attn @ tfeat; wave w owns cols [128w, 128w+128) --
    v8f d[8];
#pragma unroll
    for (int t = 0; t < 8; ++t) d[t] = (v8f){};
    for (int ks = 0; ks < TOKP / 32; ++ks) {
        v16h a = load_a_h(&s_at[ks * 32], TOKP, lane);
#pragma unroll
        for (int t = 0; t < 8; ++t) {
            v16h b = load_b_h(tfT + (long)(w * 128 + t * 16) * TOKP + ks * 32,
                              TOKP, lane);
            d[t] = wmma_f16(a, b, d[t]);
        }
    }
    // ---- epilogue: U = (delta + body) as f16 ----
    {
        int rowhi = (lane >> 4) * 8, coll = lane & 15;
#pragma unroll
        for (int t = 0; t < 8; ++t) {
            int n = w * 128 + t * 16 + coll;
#pragma unroll
            for (int r = 0; r < 8; ++r) {
                long m = qbase + rowhi + r;
                float u = d[t][r] + body[m * EMBED + n];
                U[m * EMBED + n] = (_Float16)u;
            }
        }
    }
}

// ---------------------------------------------------------------------------
// Final GEMM: Y = U @ Wf^T + bf ; out = body + scale*Y
// Block tile 64(M) x 256(N); 8 waves as 2x4; wave tile 32x64 (8 accums).
// K-slices (32 wide) double-buffered in LDS via async global->LDS copies:
//   per K-step each thread issues 5 B128 copies (1 for A, 4 for B),
//   s_wait_asynccnt 5 overlaps the next slice's copies with compute.
// Staged rows padded to LDSPAD=40 halfs -> conflict-free ds_load_b128.
// ---------------------------------------------------------------------------
__device__ inline void stage_slice(const _Float16* U, const _Float16* WfH,
                                   _Float16* sA, _Float16* sB,
                                   long mblk, int nblk, int ks, int t) {
    // A: 64 rows x 32 halfs = 256 x 16B chunks; thread t -> 1 chunk
    int arow = t >> 2, apart = (t & 3) * 8;
    async_cp16(U + (mblk + arow) * EMBED + ks * 32 + apart,
               sA + arow * LDSPAD + apart);
    // B: 256 rows x 32 halfs; thread t -> row t (4 x 16B chunks)
    const _Float16* gb = WfH + (long)(nblk + t) * EMBED + ks * 32;
    _Float16* lb = sB + t * LDSPAD;
#pragma unroll
    for (int p = 0; p < 4; ++p) async_cp16(gb + p * 8, lb + p * 8);
}

__global__ __launch_bounds__(256) void k_final(const float* x, const _Float16* U,
                                               const _Float16* WfH,
                                               const float* bf,
                                               const float* scalep,
                                               float* out) {
    __shared__ _Float16 sA[2][64 * LDSPAD];    // 2 x 5120 B
    __shared__ _Float16 sB[2][256 * LDSPAD];   // 2 x 20480 B

    const float* body = x + CLS_ELEMS;
    float* outf = out + CLS_ELEMS;
    const int tid = threadIdx.x;
    const int w = tid >> 5, lane = tid & 31;
    const int wm = w >> 2, wn = w & 3;
    const long mblk = (long)blockIdx.x * 64;
    const int  nblk = blockIdx.y * 256;

    v8f acc[2][4];
#pragma unroll
    for (int i = 0; i < 2; ++i)
#pragma unroll
        for (int t = 0; t < 4; ++t) acc[i][t] = (v8f){};

    stage_slice(U, WfH, sA[0], sB[0], mblk, nblk, 0, tid);

    for (int ks = 0; ks < EMBED / 32; ++ks) {
        const int cur = ks & 1;
        if (ks + 1 < EMBED / 32) {
            stage_slice(U, WfH, sA[cur ^ 1], sB[cur ^ 1], mblk, nblk,
                        ks + 1, tid);
            WAIT_ASYNC(5);    // current slice done; next (5 ops) in flight
        } else {
            WAIT_ASYNC(0);
        }
        __syncthreads();      // all waves' copies visible

        v16h a0 = load_a_h(&sA[cur][(wm * 32) * LDSPAD], LDSPAD, lane);
        v16h a1 = load_a_h(&sA[cur][(wm * 32 + 16) * LDSPAD], LDSPAD, lane);
#pragma unroll
        for (int t = 0; t < 4; ++t) {
            v16h b = load_b_h(&sB[cur][(wn * 64 + t * 16) * LDSPAD],
                              LDSPAD, lane);
            acc[0][t] = wmma_f16(a0, b, acc[0][t]);
            acc[1][t] = wmma_f16(a1, b, acc[1][t]);
        }
        __syncthreads();      // everyone done reading buf before overwrite
    }

    float sc = *scalep;
    int rowhi = (lane >> 4) * 8, coll = lane & 15;
#pragma unroll
    for (int i = 0; i < 2; ++i)
#pragma unroll
        for (int t = 0; t < 4; ++t) {
            int n = nblk + wn * 64 + t * 16 + coll;
            float bfn = bf[n];
#pragma unroll
            for (int r = 0; r < 8; ++r) {
                long m = mblk + wm * 32 + i * 16 + rowhi + r;
                long off = m * EMBED + n;
                outf[off] = body[off] + sc * (acc[i][t][r] + bfn);
            }
        }
}

// ---------------------------------------------------------------------------
// Launch
// ---------------------------------------------------------------------------
extern "C" void kernel_launch(void* const* d_in, const int* in_sizes, int n_in,
                              void* d_out, int out_size, void* d_ws,
                              size_t ws_size, hipStream_t stream) {
    const float* x     = (const float*)d_in[0];
    const int*   layer = (const int*)  d_in[1];
    const float* A     = (const float*)d_in[2];
    const float* B     = (const float*)d_in[3];
    const float* Wd    = (const float*)d_in[4];
    const float* bd    = (const float*)d_in[5];
    const float* Wu    = (const float*)d_in[6];
    const float* bu    = (const float*)d_in[7];
    const float* Wt    = (const float*)d_in[8];
    const float* bt    = (const float*)d_in[9];
    const float* Wf    = (const float*)d_in[10];
    const float* bf    = (const float*)d_in[11];
    const float* scale = (const float*)d_in[12];
    float* out = (float*)d_out;
    char*  ws  = (char*)d_ws;

    // workspace layout (all offsets 256B-aligned)
    const size_t OFF_H    = 0;                        // 1024*128*4  = 512KB
    const size_t OFF_FB   = OFF_H    + 524288;        // 1024*1024*4 = 4MB
    const size_t OFF_TOKF = OFF_FB   + 4194304;       // 128*1024*4  = 512KB
    const size_t OFF_TOKH = OFF_TOKF + 524288;        // 128*1024*2  = 256KB
    const size_t OFF_TFT  = OFF_TOKH + 262144;        // 1024*128*2  = 256KB
    const size_t OFF_WFH  = OFF_TFT  + 262144;        // 1024*1024*2 = 2MB
    const size_t OFF_U    = OFF_WFH  + 2097152;       // 32768*1024*2= 64MB
    const size_t WS_NEED  = OFF_U    + 67108864;
    if (ws_size < WS_NEED) return;

    float*     H    = (float*)    (ws + OFF_H);
    float*     fB   = (float*)    (ws + OFF_FB);
    float*     tokF = (float*)    (ws + OFF_TOKF);
    _Float16*  tokH = (_Float16*) (ws + OFF_TOKH);
    _Float16*  tfT  = (_Float16*) (ws + OFF_TFT);
    _Float16*  WfH  = (_Float16*) (ws + OFF_WFH);
    _Float16*  U    = (_Float16*) (ws + OFF_U);

    k_copy_cls<<<(CLS_ELEMS + 255) / 256, 256, 0, stream>>>(x, out);
    k_mlp1 <<<(EMBED * MLPDIM) / 256, 256, 0, stream>>>(B, layer, Wd, bd, H);
    k_mlp2 <<<(EMBED * EMBED) / 256, 256, 0, stream>>>(H, Wu, bu, fB);
    k_tokens<<<(TOKP * EMBED) / 256, 256, 0, stream>>>(A, fB, tokF, tokH);
    k_tfeat <<<(EMBED * TOKP) / 256, 256, 0, stream>>>(tokF, Wt, bt, tfT);
    k_cast  <<<(EMBED * EMBED) / 256, 256, 0, stream>>>(Wf, WfH);
    k_attn  <<<MROWS / 16, 256, 0, stream>>>(x, tokH, tfT, U);
    k_final <<<dim3(MROWS / 64, EMBED / 256), 256, 0, stream>>>(x, U, WfH, bf,
                                                               scale, out);
}
